// SchNet_var1_12799002542249
// MI455X (gfx1250) — matrix operand order repaired and checked by reading, not compiled, not run on hardware
//
#include <hip/hip_runtime.h>
#include <hip/hip_bf16.h>
#include <math.h>

typedef __bf16 v16bf __attribute__((ext_vector_type(16)));
typedef float  v8f   __attribute__((ext_vector_type(8)));

#define NHF     128
#define NRBF    50
#define NLAYER  3
#define NNODES  30
#define NMOL    1024
#define NATOMS  (NMOL*NNODES)          // 30720
#define EPM     (NNODES*(NNODES-1))    // 870 edges per molecule
#define NREACT  256
#define CUTOFF_F 5.0f
#define LOG2_F   0.6931471805599453f
#define LOG2E_F  1.4426950408889634f
#define PI_F     3.14159265358979323846f

// ---------------------------------------------------------------- helpers
// shifted softplus with raw hardware transcendentals:
//   ssp(x) = max(x,0) + ln(1+e^-|x|) - ln2
//          = max(x,0) + ln2 * (log2(1 + 2^(-|x|*log2e)) - 1)
// 1+t is in (1,2] -> no denorm/overflow fixups needed around v_exp/v_log.
__device__ __forceinline__ float sspf(float x) {
    float t = __builtin_amdgcn_exp2f(-fabsf(x) * LOG2E_F);
    float u = __builtin_amdgcn_logf(1.f + t);          // log2(1+t)
    return fmaxf(x, 0.f) + LOG2_F * (u - 1.f);
}

__device__ __forceinline__ v8f wmma_bf16(v16bf a, v16bf b, v8f c) {
    return __builtin_amdgcn_wmma_f32_16x16x32_bf16(
        /*neg_a=*/false, a, /*neg_b=*/false, b,
        /*c_mod=*/(short)0, c, /*reuse_a=*/false, /*reuse_b=*/false);
}

__device__ __forceinline__ float rbf_val(float r, int k) {
    if (k >= NRBF) return 0.f;           // zero padding K 50..63
    const float delta = CUTOFF_F / (float)(NRBF - 1);
    const float coeff = -0.5f / (delta * delta);
    float d = r - (float)k * delta;
    return __expf(coeff * d * d);
}

// scheduling fence: stops the compiler from mass-hoisting fragment loads
// across nt-iterations (caused scratch spills in round 1)
#define SCHED_FENCE() asm volatile("" ::: "memory")

// ---------------------------------------------------------------- kernels
__global__ void zero_kernel(float* p, int n) {
    int i = blockIdx.x * blockDim.x + threadIdx.x;
    if (i < n) p[i] = 0.f;
}

__global__ void init_h_kernel(const int* __restrict__ z, const float* __restrict__ emb,
                              float* __restrict__ h) {
    int idx = blockIdx.x * blockDim.x + threadIdx.x;
    if (idx >= NATOMS * NHF) return;
    int a = idx >> 7, f = idx & 127;
    h[idx] = emb[z[a] * NHF + f];
}

// Pack a K x 128 row-major fp32 weight matrix into per-lane B-fragment layout:
// out[((kt*8+nt)*32+lane)*16+j] = W[kt*32 + (lane>=16?8:0) + (j&7) + (j>=8?16:0)][nt*16 + (lane&15)]
__global__ void pack_weight_kernel(const float* __restrict__ W, __bf16* __restrict__ out,
                                   int K, int ktiles) {
    int idx = blockIdx.x * blockDim.x + threadIdx.x;
    int total = ktiles * 8 * 32 * 16;
    if (idx >= total) return;
    int j    = idx & 15;
    int lane = (idx >> 4) & 31;
    int nt   = (idx >> 9) & 7;
    int kt   = idx >> 12;
    int col = nt * 16 + (lane & 15);
    int k   = kt * 32 + ((lane >> 4) ? 8 : 0) + (j & 7) + ((j >> 3) ? 16 : 0);
    float v = (k < K) ? W[k * NHF + col] : 0.f;
    out[idx] = (__bf16)v;
}

// Y = opt_ssp(X @ W + b), X:[NATOMS,128], W packed 4x8 frags, 256 thr = 8 waves,
// one 16-row tile per wave. OUT_ADD: Yf += result (fp32); else Ybf = bf16(result).
template <typename TIN, bool DO_SSP, bool OUT_ADD>
__global__ void atom_gemm_kernel(const TIN* __restrict__ X, const __bf16* __restrict__ Wp,
                                 const float* __restrict__ bias,
                                 __bf16* __restrict__ Ybf, float* __restrict__ Yf) {
    __shared__ __attribute__((aligned(32))) __bf16 ldsW[4 * 8 * 32 * 16]; // 32 KB
    __shared__ float ldsB[NHF];
    int tid = threadIdx.x;
    { // cooperative weight staging (16384 bf16 = 2048 uint4)
        const uint4* src = (const uint4*)Wp;
        uint4* dst = (uint4*)ldsW;
        for (int i = tid; i < (4 * 8 * 32 * 16) / 8; i += 256) dst[i] = src[i];
        for (int i = tid; i < NHF; i += 256) ldsB[i] = bias ? bias[i] : 0.f;
    }
    __syncthreads();

    int wave = tid >> 5, lane = tid & 31;
    int m = lane & 15, half = lane >> 4, mb = half * 8, n = m;
    int rowbase = blockIdx.x * 128 + wave * 16;
    int row = rowbase + m;

    v16bf afrag[4];
#pragma unroll
    for (int kt = 0; kt < 4; ++kt) {
        const TIN* p = X + (size_t)row * NHF + kt * 32 + half * 8;
#pragma unroll
        for (int j = 0; j < 8; ++j) {
            afrag[kt][j]     = (__bf16)(float)p[j];
            afrag[kt][8 + j] = (__bf16)(float)p[16 + j];
        }
    }
#pragma unroll
    for (int nt = 0; nt < 8; ++nt) {
        SCHED_FENCE();
        float bv = ldsB[nt * 16 + n];
        v8f acc;
#pragma unroll
        for (int v = 0; v < 8; ++v) acc[v] = bv;
#pragma unroll
        for (int kt = 0; kt < 4; ++kt) {
            const v16bf* bp = (const v16bf*)&ldsW[((kt * 8 + nt) * 32 + lane) * 16];
            acc = wmma_bf16(afrag[kt], *bp, acc);
        }
#pragma unroll
        for (int v = 0; v < 8; ++v) {
            float val = acc[v];
            if (DO_SSP) val = sspf(val);
            size_t oi = (size_t)(rowbase + mb + v) * NHF + nt * 16 + n;
            if (OUT_ADD) Yf[oi] += val;
            else         Ybf[oi] = (__bf16)val;
        }
    }
}

// One workgroup per molecule. Edge tiles are per-SOURCE-ROW: row r (29 real
// edges, padded to 32) -> 2 half-tiles of 16 edges, all scattering to atom r.
// Per tile: rbf -> GEMM1(WMMA) -> ssp -> GEMM2(WMMA) -> *dist_cut *xf[col]
// -> in-register sum over the 8 C-rows -> ONE ds_add_f32 per (lane,nt).
__global__ void edge_kernel(const float* __restrict__ coord,
                            const float* __restrict__ edge_mask,
                            const __bf16* __restrict__ xf,
                            const __bf16* __restrict__ W1p,  // [2][8][32][16]
                            const __bf16* __restrict__ W2p,  // [4][8][32][16]
                            const float* __restrict__ b1,
                            const float* __restrict__ b2,
                            float* __restrict__ mout) {
    __shared__ float c_lds[NNODES][3];
    __shared__ __attribute__((aligned(32))) __bf16 xf_lds[NNODES][NHF];     // 7.5 KB
    __shared__ float m_lds[NNODES][NHF];                                     // 15 KB
    __shared__ __attribute__((aligned(32))) __bf16 t_lds[8][16][NHF];        // 32 KB
    __shared__ __attribute__((aligned(32))) __bf16 w1_lds[2 * 8 * 32 * 16]; // 16 KB
    __shared__ __attribute__((aligned(32))) __bf16 w2_lds[4 * 8 * 32 * 16]; // 32 KB
    __shared__ float b1_lds[NHF], b2_lds[NHF];

    int tid = threadIdx.x, b = blockIdx.x;
    int wave = tid >> 5, lane = tid & 31;

    for (int i = tid; i < NNODES * 3; i += 256) ((float*)c_lds)[i] = coord[b * NNODES * 3 + i];
    {
        const uint* src = (const uint*)(xf + (size_t)b * NNODES * NHF);
        uint* dst = (uint*)xf_lds;
        for (int i = tid; i < NNODES * NHF / 2; i += 256) dst[i] = src[i];
    }
    { // weight fragment staging
        const uint4* s1 = (const uint4*)W1p;  uint4* d1 = (uint4*)w1_lds;
        for (int i = tid; i < (2 * 8 * 32 * 16) / 8; i += 256) d1[i] = s1[i];
        const uint4* s2 = (const uint4*)W2p;  uint4* d2 = (uint4*)w2_lds;
        for (int i = tid; i < (4 * 8 * 32 * 16) / 8; i += 256) d2[i] = s2[i];
    }
    for (int i = tid; i < NNODES * NHF; i += 256) ((float*)m_lds)[i] = 0.f;
    for (int i = tid; i < NHF; i += 256) { b1_lds[i] = b1[i]; b2_lds[i] = b2[i]; }
    __syncthreads();

    int m = lane & 15, half = lane >> 4, mb = half * 8, n = m;
    const int NT = NNODES * 2;  // 60 row-half tiles

    for (int tile = wave; tile < NT; tile += 8) {
        int rrow = tile >> 1;          // target atom (segment row)
        int cb   = (tile & 1) * 16;    // neighbor-slot base: c' in [cb, cb+16)

        // this lane's A-fragment edge: neighbor slot c' = cb + m
        int cp = cb + m;
        int cA = cp + (cp >= rrow ? 1 : 0);
        bool vA = cp < (NNODES - 1);
        int colA = vA ? cA : 0;
        float dx = c_lds[rrow][0] - c_lds[colA][0];
        float dy = c_lds[rrow][1] - c_lds[colA][1];
        float dz = c_lds[rrow][2] - c_lds[colA][2];
        float r = sqrtf(dx * dx + dy * dy + dz * dz);

        // A-fragments for GEMM1 built from on-the-fly Gaussian RBF
        v16bf a1[2];
#pragma unroll
        for (int kt = 0; kt < 2; ++kt) {
#pragma unroll
            for (int j = 0; j < 8; ++j) {
                int k1 = kt * 32 + half * 8 + j;
                a1[kt][j]     = (__bf16)(vA ? rbf_val(r, k1) : 0.f);
                a1[kt][8 + j] = (__bf16)(vA ? rbf_val(r, k1 + 16) : 0.f);
            }
        }

        // per-C-row metadata (rows mb..mb+7 -> neighbor slots cb+mb+v)
        float dcv[8];
        int cl[8];
#pragma unroll
        for (int v = 0; v < 8; ++v) {
            int cpv = cb + mb + v;
            int cv = cpv + (cpv >= rrow ? 1 : 0);
            bool ok = cpv < (NNODES - 1);
            cl[v] = ok ? cv : 0;
            float ddx = c_lds[rrow][0] - c_lds[cl[v]][0];
            float ddy = c_lds[rrow][1] - c_lds[cl[v]][1];
            float ddz = c_lds[rrow][2] - c_lds[cl[v]][2];
            float rv = sqrtf(ddx * ddx + ddy * ddy + ddz * ddz);
            float dc = 0.5f * (__cosf(rv * (PI_F / CUTOFF_F)) + 1.f);
            dc = (rv < CUTOFF_F) ? dc : 0.f;
            int ge = rrow * (NNODES - 1) + (ok ? cpv : 0);
            dcv[v] = ok ? dc * edge_mask[(size_t)b * EPM + ge] : 0.f;
        }

        // GEMM1: t = ssp(rbf @ f_W1 + b1)  -> t_lds (bf16, row-major 16x128)
#pragma unroll
        for (int nt = 0; nt < 8; ++nt) {
            SCHED_FENCE();
            float bv = b1_lds[nt * 16 + n];
            v8f acc;
#pragma unroll
            for (int v = 0; v < 8; ++v) acc[v] = bv;
#pragma unroll
            for (int kt = 0; kt < 2; ++kt) {
                const v16bf* bp = (const v16bf*)&w1_lds[((kt * 8 + nt) * 32 + lane) * 16];
                acc = wmma_bf16(a1[kt], *bp, acc);
            }
#pragma unroll
            for (int v = 0; v < 8; ++v)
                t_lds[wave][mb + v][nt * 16 + n] = (__bf16)sspf(acc[v]);
        }
        // same-wave LDS RAW: C-layout stores above, A-layout loads below
        asm volatile("s_wait_dscnt 0x0" ::: "memory");

        // A-fragments for GEMM2 from t_lds
        v16bf a2[4];
#pragma unroll
        for (int kt = 0; kt < 4; ++kt) {
            const __bf16* p = &t_lds[wave][m][kt * 32 + half * 8];
#pragma unroll
            for (int j = 0; j < 8; ++j) {
                a2[kt][j]     = p[j];
                a2[kt][8 + j] = p[16 + j];
            }
        }

        // GEMM2: Wij = (t @ f_W2 + b2) * dist_cut ; x_ij = xf[col]*Wij
        // all 16 C-rows scatter to atom rrow -> reduce over v, one atomic.
#pragma unroll
        for (int nt = 0; nt < 8; ++nt) {
            SCHED_FENCE();
            float bv = b2_lds[nt * 16 + n];
            v8f acc;
#pragma unroll
            for (int v = 0; v < 8; ++v) acc[v] = bv;
#pragma unroll
            for (int kt = 0; kt < 4; ++kt) {
                const v16bf* bp = (const v16bf*)&w2_lds[((kt * 8 + nt) * 32 + lane) * 16];
                acc = wmma_bf16(a2[kt], *bp, acc);
            }
            float s = 0.f;
#pragma unroll
            for (int v = 0; v < 8; ++v)
                s += acc[v] * dcv[v] * (float)xf_lds[cl[v]][nt * 16 + n];
            atomicAdd(&m_lds[rrow][nt * 16 + n], s);
        }
    }
    __syncthreads();
    float* mo = mout + (size_t)b * NNODES * NHF;
    for (int i = tid; i < NNODES * NHF; i += 256) mo[i] = ((float*)m_lds)[i];
}

// Per-molecule readout: sum atoms, decoder MLP, scatter-add to reactions.
__global__ void readout_kernel(const float* __restrict__ h, const float* __restrict__ amask,
                               const int* __restrict__ ridx, const float* __restrict__ rsign,
                               const float* __restrict__ dW1, const float* __restrict__ db1,
                               const float* __restrict__ dW2, const float* __restrict__ db2,
                               float* __restrict__ pred) {
    __shared__ float hs[NHF];
    __shared__ float d1[NHF / 2];
    int b = blockIdx.x, tid = threadIdx.x;  // 128 threads
    float s = 0.f;
    for (int a = 0; a < NNODES; ++a)
        s += h[((size_t)b * NNODES + a) * NHF + tid] * amask[b * NNODES + a];
    hs[tid] = s;
    __syncthreads();
    if (tid < NHF / 2) {
        float acc = db1[tid];
        for (int f = 0; f < NHF; ++f) acc += hs[f] * dW1[f * (NHF / 2) + tid];
        d1[tid] = sspf(acc);
    }
    __syncthreads();
    if (tid == 0) {
        float acc = db2[0];
        for (int f = 0; f < NHF / 2; ++f) acc += d1[f] * dW2[f];
        atomicAdd(&pred[ridx[b]], acc * rsign[b]);
    }
}

// ---------------------------------------------------------------- launch
extern "C" void kernel_launch(void* const* d_in, const int* in_sizes, int n_in,
                              void* d_out, int out_size, void* d_ws, size_t ws_size,
                              hipStream_t stream) {
    (void)in_sizes; (void)n_in; (void)out_size; (void)ws_size;
    const int*   z      = (const int*)d_in[0];
    const float* coord  = (const float*)d_in[1];
    const float* amask  = (const float*)d_in[3];
    const float* emask  = (const float*)d_in[4];
    const int*   ridx   = (const int*)d_in[5];
    const float* rsign  = (const float*)d_in[6];
    const float* emb    = (const float*)d_in[7];
    const float* in2f_W = (const float*)d_in[10];
    const float* f_W1   = (const float*)d_in[11];
    const float* f_b1   = (const float*)d_in[12];
    const float* f_W2   = (const float*)d_in[13];
    const float* f_b2   = (const float*)d_in[14];
    const float* o_W1   = (const float*)d_in[15];
    const float* o_b1   = (const float*)d_in[16];
    const float* o_W2   = (const float*)d_in[17];
    const float* o_b2   = (const float*)d_in[18];
    const float* dW1    = (const float*)d_in[19];
    const float* db1    = (const float*)d_in[20];
    const float* dW2    = (const float*)d_in[21];
    const float* db2    = (const float*)d_in[22];

    // workspace carve-out
    char* w = (char*)d_ws;
    size_t off = 0;
    float* h = (float*)(w + off);       off += (size_t)NATOMS * NHF * 4;   // 15.7 MB
    float* mbuf = (float*)(w + off);    off += (size_t)NATOMS * NHF * 4;   // 15.7 MB
    __bf16* xf = (__bf16*)(w + off);    off += (size_t)NATOMS * NHF * 2;   // 7.9 MB
    __bf16* tbuf = (__bf16*)(w + off);  off += (size_t)NATOMS * NHF * 2;   // 7.9 MB
    const size_t P128 = 4 * 8 * 32 * 16;  // packed 128x128: 16384 bf16
    const size_t P64  = 2 * 8 * 32 * 16;  // packed 64x128:   8192 bf16
    __bf16* in2f_p = (__bf16*)(w + off); off += NLAYER * P128 * 2;
    __bf16* fW1_p  = (__bf16*)(w + off); off += NLAYER * P64 * 2;
    __bf16* fW2_p  = (__bf16*)(w + off); off += NLAYER * P128 * 2;
    __bf16* oW1_p  = (__bf16*)(w + off); off += NLAYER * P128 * 2;
    __bf16* oW2_p  = (__bf16*)(w + off); off += NLAYER * P128 * 2;

    zero_kernel<<<1, 256, 0, stream>>>((float*)d_out, NREACT);

    for (int l = 0; l < NLAYER; ++l) {
        pack_weight_kernel<<<(int)(P128 + 255) / 256, 256, 0, stream>>>(
            in2f_W + (size_t)l * NHF * NHF, in2f_p + l * P128, NHF, 4);
        pack_weight_kernel<<<(int)(P64 + 255) / 256, 256, 0, stream>>>(
            f_W1 + (size_t)l * NRBF * NHF, fW1_p + l * P64, NRBF, 2);
        pack_weight_kernel<<<(int)(P128 + 255) / 256, 256, 0, stream>>>(
            f_W2 + (size_t)l * NHF * NHF, fW2_p + l * P128, NHF, 4);
        pack_weight_kernel<<<(int)(P128 + 255) / 256, 256, 0, stream>>>(
            o_W1 + (size_t)l * NHF * NHF, oW1_p + l * P128, NHF, 4);
        pack_weight_kernel<<<(int)(P128 + 255) / 256, 256, 0, stream>>>(
            o_W2 + (size_t)l * NHF * NHF, oW2_p + l * P128, NHF, 4);
    }

    init_h_kernel<<<(NATOMS * NHF + 255) / 256, 256, 0, stream>>>(z, emb, h);

    const int GB = NATOMS / 128;  // 240 blocks for atom GEMMs
    for (int l = 0; l < NLAYER; ++l) {
        // xf = h @ in2f_W[l]   (no bias, no act) -> bf16
        atom_gemm_kernel<float, false, false><<<GB, 256, 0, stream>>>(
            h, in2f_p + l * P128, nullptr, xf, nullptr);
        // fused edge pipeline -> m
        edge_kernel<<<NMOL, 256, 0, stream>>>(
            coord, emask, xf, fW1_p + l * P64, fW2_p + l * P128,
            f_b1 + l * NHF, f_b2 + l * NHF, mbuf);
        // t = ssp(m @ o_W1 + o_b1) -> bf16
        atom_gemm_kernel<float, true, false><<<GB, 256, 0, stream>>>(
            mbuf, oW1_p + l * P128, o_b1 + l * NHF, tbuf, nullptr);
        // h += t @ o_W2 + o_b2
        atom_gemm_kernel<__bf16, false, true><<<GB, 256, 0, stream>>>(
            tbuf, oW2_p + l * P128, o_b2 + l * NHF, nullptr, h);
    }

    readout_kernel<<<NMOL, 128, 0, stream>>>(h, amask, ridx, rsign, dW1, db1, dW2, db2,
                                             (float*)d_out);
}
